// IPA_53944789238386
// MI455X (gfx1250) — matrix operand order repaired and checked by reading, not compiled, not run on hardware
//
#include <hip/hip_runtime.h>
#include <hip/hip_bf16.h>
#include <math.h>

// ---------------------------------------------------------------------------
// IPA (Invariant Point Attention) for gfx1250 / MI455X.
// Dense projections use V_WMMA_F32_16X16X32_F16 on pre-converted f16 operands;
// edge ops use VALU + atomics.
// ---------------------------------------------------------------------------

typedef __attribute__((ext_vector_type(16))) _Float16 v16h;
typedef __attribute__((ext_vector_type(8)))  float    v8f;

#define GEMM_BM 128
#define GEMM_BN 128
#define GEMM_BK 32

union Frag16 { v16h v; unsigned int u[8]; };
union Acc8   { v8f  v; float f[8]; };

// ---------------------------------------------------------------------------
// f32 -> f16 bulk conversion, 8 elements / thread (n must be multiple of 8).
__global__ void ipa_cvt_f16(const float* __restrict__ src, _Float16* __restrict__ dst,
                            long long n8)
{
    long long idx = (long long)blockIdx.x * blockDim.x + threadIdx.x;
    if (idx >= n8) return;
    const float4* s4 = (const float4*)(src + idx * 8);
    float4 a = s4[0], b = s4[1];
    union { _Float16 h[8]; uint4 u; } t;
    t.h[0] = (_Float16)a.x; t.h[1] = (_Float16)a.y;
    t.h[2] = (_Float16)a.z; t.h[3] = (_Float16)a.w;
    t.h[4] = (_Float16)b.x; t.h[5] = (_Float16)b.y;
    t.h[6] = (_Float16)b.z; t.h[7] = (_Float16)b.w;
    *(uint4*)(dst + idx * 8) = t.u;
}

// ---------------------------------------------------------------------------
// C[M x Ncols] = A[M x K] * W[Ncols x K]^T + bias[Ncols]   (A, W are f16)
// K % 32 == 0 (all K here are 384/128/768).
// Out-of-range rows/cols handled by address clamping only: clamped A rows feed
// only C rows >= M (never stored); clamped B rows feed only C cols >= Ncols
// (never stored). Staging uses register double-buffering: tile t+1's global
// loads are issued before tile t's WMMA block, so HBM latency hides behind
// compute and the loadcnt wait lands at the next iteration's LDS stores.
// 8 waves: 4 row-groups x 2 col-groups; each wave owns 32x64 = 2x4 WMMA tiles.
__global__ __launch_bounds__(256)
void ipa_gemm_wmma(const _Float16* __restrict__ A, const _Float16* __restrict__ W,
                   const float* __restrict__ bias, float* __restrict__ C,
                   int M, int Ncols, int K)
{
    __shared__ _Float16 As[GEMM_BM * GEMM_BK];
    __shared__ _Float16 Bs[GEMM_BN * GEMM_BK];

    const int tid   = threadIdx.x;
    const int lane  = tid & 31;
    const int wave  = tid >> 5;      // 0..7
    const int hi    = lane >> 4;     // 0/1 (half-wave)
    const int ln    = lane & 15;
    const int waveM = wave & 3;      // 4 row groups of 32
    const int waveN = wave >> 2;     // 2 col groups of 64
    const int rowBase = blockIdx.x * GEMM_BM;
    const int colBase = blockIdx.y * GEMM_BN;

    // ---- loop-invariant staging indices (8-half = 16B chunks) ----
    // Each tile is 128x32 halves = 512 chunks -> 2 per thread.
    size_t aOff[2], bOff[2];
    int aLds[2], bLds[2];
#pragma unroll
    for (int it = 0; it < 2; ++it) {
        int chunk = it * 256 + tid;
        int r  = chunk >> 2;           // 0..127
        int cg = (chunk & 3) << 3;     // 0,8,16,24
        int gm = rowBase + r; if (gm > M - 1) gm = M - 1;
        int gn = colBase + r; if (gn > Ncols - 1) gn = Ncols - 1;
        aOff[it] = (size_t)gm * K + cg;
        bOff[it] = (size_t)gn * K + cg;
        aLds[it] = r * GEMM_BK + cg;
        bLds[it] = r * GEMM_BK + cg;
    }

    Acc8 acc[2][4];
#pragma unroll
    for (int i = 0; i < 2; ++i)
#pragma unroll
        for (int j = 0; j < 4; ++j)
#pragma unroll
            for (int r = 0; r < 8; ++r) acc[i][j].f[r] = 0.0f;

    // ---- prologue: load tile 0 into registers ----
    uint4 ra0 = *(const uint4*)&A[aOff[0]];
    uint4 ra1 = *(const uint4*)&A[aOff[1]];
    uint4 rb0 = *(const uint4*)&W[bOff[0]];
    uint4 rb1 = *(const uint4*)&W[bOff[1]];

    for (int k0 = 0; k0 < K; k0 += GEMM_BK) {
        // ---- commit current tile to LDS ----
        *(uint4*)&As[aLds[0]] = ra0;
        *(uint4*)&As[aLds[1]] = ra1;
        *(uint4*)&Bs[bLds[0]] = rb0;
        *(uint4*)&Bs[bLds[1]] = rb1;
        // ---- issue next tile's loads now; they complete under the WMMAs ----
        int kn = k0 + GEMM_BK;
        if (kn < K) {
            ra0 = *(const uint4*)&A[aOff[0] + kn];
            ra1 = *(const uint4*)&A[aOff[1] + kn];
            rb0 = *(const uint4*)&W[bOff[0] + kn];
            rb1 = *(const uint4*)&W[bOff[1] + kn];
            if (kn + GEMM_BK < K) {
                __builtin_prefetch(&A[aOff[0] + kn + GEMM_BK], 0, 2);
                __builtin_prefetch(&W[bOff[0] + kn + GEMM_BK], 0, 2);
            }
        }
        __syncthreads();

        // ---- assemble fragments per ISA 7.12.2 layouts ----
        Frag16 afrag[2], bfrag[4];
#pragma unroll
        for (int ti = 0; ti < 2; ++ti) {
            const _Float16* Arow = &As[(waveM * 32 + ti * 16 + ln) * GEMM_BK];
#pragma unroll
            for (int vv = 0; vv < 8; ++vv) {
                int kk = ((vv >> 2) << 4) + hi * 8 + ((vv & 3) << 1); // K pair base
                afrag[ti].u[vv] = *(const unsigned int*)&Arow[kk];
            }
        }
#pragma unroll
        for (int tj = 0; tj < 4; ++tj) {
            const _Float16* Brow = &Bs[(waveN * 64 + tj * 16 + ln) * GEMM_BK];
#pragma unroll
            for (int vv = 0; vv < 8; ++vv) {
                bfrag[tj].u[vv] = *(const unsigned int*)&Brow[((vv >> 2) << 4) + hi * 8 + ((vv & 3) << 1)];
            }
        }

        // ---- 8x V_WMMA_F32_16X16X32_F16 ----
#pragma unroll
        for (int ti = 0; ti < 2; ++ti)
#pragma unroll
            for (int tj = 0; tj < 4; ++tj)
                acc[ti][tj].v = __builtin_amdgcn_wmma_f32_16x16x32_f16(
                    false, afrag[ti].v, false, bfrag[tj].v,
                    (short)0, acc[ti][tj].v, false, false);
        __syncthreads();
    }

    // ---- store C + bias (C/D layout: lane n = l&15, row = hi*8 + vgpr) ----
#pragma unroll
    for (int ti = 0; ti < 2; ++ti) {
#pragma unroll
        for (int tj = 0; tj < 4; ++tj) {
            int n = colBase + waveN * 64 + tj * 16 + ln;
            if (n < Ncols) {
                float bv = bias[n];
#pragma unroll
                for (int r = 0; r < 8; ++r) {
                    int m = rowBase + waveM * 32 + ti * 16 + hi * 8 + r;
                    if (m < M) C[(size_t)m * Ncols + n] = acc[ti][tj].f[r] + bv;
                }
            }
        }
    }
}

// ---------------------------------------------------------------------------
// pts <- R*pts + t (in place).  pts layout [n][npts][3], f layout [n][4][4].
__global__ void ipa_pts_to_global(const float* __restrict__ f, float* __restrict__ pts,
                                  int npts, long long total)
{
    long long idx = (long long)blockIdx.x * blockDim.x + threadIdx.x;
    if (idx >= total) return;
    long long n = idx / npts;
    long long p = idx - n * npts;
    const float* fr = &f[n * 16];
    float* pv = &pts[(n * npts + p) * 3];
    float x = pv[0], y = pv[1], z = pv[2];
    pv[0] = fr[0] * x + fr[1] * y + fr[2]  * z + fr[3];
    pv[1] = fr[4] * x + fr[5] * y + fr[6]  * z + fr[7];
    pv[2] = fr[8] * x + fr[9] * y + fr[10] * z + fr[11];
}

// ---------------------------------------------------------------------------
__global__ void ipa_fill(float* __restrict__ p, float val, long long n)
{
    long long idx = (long long)blockIdx.x * blockDim.x + threadIdx.x;
    if (idx < n) p[idx] = val;
}

// ---------------------------------------------------------------------------
// attention logits per (edge, head); vectorized gathers (rows are 16/12-float
// aligned so float4 loads are legal).
__global__ void ipa_edge_logits(const int* __restrict__ ei,
                                const float* __restrict__ q, const float* __restrict__ k,
                                const float* __restrict__ qp, const float* __restrict__ kp,
                                const float* __restrict__ biasz, const float* __restrict__ head_w,
                                float* __restrict__ attn, int Ecnt)
{
    int idx = blockIdx.x * blockDim.x + threadIdx.x;
    if (idx >= Ecnt * 12) return;
    int e = idx / 12;
    int h = idx - e * 12;
    int i = ei[2 * e], j = ei[2 * e + 1];

    const float4* qi = (const float4*)&q[(size_t)i * 192 + h * 16];
    const float4* kj = (const float4*)&k[(size_t)j * 192 + h * 16];
    float dot = 0.0f;
#pragma unroll
    for (int d = 0; d < 4; ++d) {
        float4 a = qi[d], b = kj[d];
        dot += a.x * b.x + a.y * b.y + a.z * b.z + a.w * b.w;
    }
    float att = dot * 6.928203230275509f;                         // sqrt(3*D)
    att += biasz[(size_t)e * 12 + h] * 1.7320508075688772f;       // sqrt(3)

    const float4* qpi = (const float4*)&qp[(size_t)i * 144 + h * 12];
    const float4* kpj = (const float4*)&kp[(size_t)j * 144 + h * 12];
    float ss = 0.0f;
#pragma unroll
    for (int u = 0; u < 3; ++u) {
        float4 a = qpi[u], b = kpj[u];
        float d0 = a.x - b.x, d1 = a.y - b.y, d2 = a.z - b.z, d3 = a.w - b.w;
        ss += d0 * d0 + d1 * d1 + d2 * d2 + d3 * d3;
    }
    float hw = head_w[h];
    float sp = (hw > 20.0f) ? hw : log1pf(__expf(hw));            // softplus
    att += ss * (sp * -3.6742346141747673f);                      // -0.5*sqrt(54)
    attn[idx] = att;
}

// ---------------------------------------------------------------------------
__device__ inline void atomicMaxFloat(float* addr, float value)
{
    if (value >= 0.0f)
        atomicMax((int*)addr, __float_as_int(value));
    else
        atomicMin((unsigned int*)addr, __float_as_uint(value));
}

__global__ void ipa_seg_max(const int* __restrict__ ei, const float* __restrict__ attn,
                            float* __restrict__ m, int Ecnt)
{
    int idx = blockIdx.x * blockDim.x + threadIdx.x;
    if (idx >= Ecnt * 12) return;
    int e = idx / 12;
    int h = idx - e * 12;
    int i = ei[2 * e];
    atomicMaxFloat(&m[(size_t)i * 12 + h], attn[idx]);
}

// ex = exp(attn - m[i]);  denom[i] += ex   (ex overwrites attn in place)
__global__ void ipa_exp_denom(const int* __restrict__ ei, float* __restrict__ attn,
                              const float* __restrict__ m, float* __restrict__ denom, int Ecnt)
{
    int idx = blockIdx.x * blockDim.x + threadIdx.x;
    if (idx >= Ecnt * 12) return;
    int e = idx / 12;
    int h = idx - e * 12;
    int i = ei[2 * e];
    float ex = __expf(attn[idx] - m[(size_t)i * 12 + h]);
    attn[idx] = ex;
    atomicAdd(&denom[(size_t)i * 12 + h], ex);
}

// ---------------------------------------------------------------------------
// scatter-aggregate o / o_pts_global / o_pair.
// 672 channels per edge = 84 blocks of 8 contiguous channels; blocks never
// straddle a head (192, 288, 192 all divisible by 8; 8 | 16 and 8 | 24).
__global__ void ipa_aggregate(const int* __restrict__ ei, const float* __restrict__ ex,
                              const float* __restrict__ denom,
                              const float* __restrict__ v, const float* __restrict__ vp,
                              const float* __restrict__ vpair,
                              float* __restrict__ o, float* __restrict__ og,
                              float* __restrict__ opair, long long total)
{
    long long idx = (long long)blockIdx.x * blockDim.x + threadIdx.x;
    if (idx >= total) return;
    int cb = (int)(idx % 84);
    long long e = idx / 84;
    int i = ei[2 * e], j = ei[2 * e + 1];

    const float* src;
    float* dst;
    int h;
    if (cb < 24) {
        int coff = cb << 3;            // 0..184
        h = coff >> 4;
        src = &v[(size_t)j * 192 + coff];
        dst = &o[(size_t)i * 192 + coff];
    } else if (cb < 60) {
        int coff = (cb - 24) << 3;     // 0..280
        h = coff / 24;                 // PV*3 = 24 per head
        src = &vp[(size_t)j * 288 + coff];
        dst = &og[(size_t)i * 288 + coff];
    } else {
        int coff = (cb - 60) << 3;     // 0..184
        h = coff >> 4;
        src = &vpair[(size_t)e * 192 + coff];
        dst = &opair[(size_t)i * 192 + coff];
    }
    float a = ex[e * 12 + h] / denom[(size_t)i * 12 + h];
    float4 s0 = *(const float4*)&src[0];
    float4 s1 = *(const float4*)&src[4];
    atomicAdd(&dst[0], a * s0.x);
    atomicAdd(&dst[1], a * s0.y);
    atomicAdd(&dst[2], a * s0.z);
    atomicAdd(&dst[3], a * s0.w);
    atomicAdd(&dst[4], a * s1.x);
    atomicAdd(&dst[5], a * s1.y);
    atomicAdd(&dst[6], a * s1.z);
    atomicAdd(&dst[7], a * s1.w);
}

// ---------------------------------------------------------------------------
// build o_s[N x 768] (f16, feeds final WMMA GEMM) =
//   [o | R^T(o_pts_g - t) | ||.|| | o_pair]
__global__ void ipa_finalize(const float* __restrict__ f, const float* __restrict__ o,
                             const float* __restrict__ og, const float* __restrict__ opair,
                             _Float16* __restrict__ os, long long total)
{
    long long idx = (long long)blockIdx.x * blockDim.x + threadIdx.x;
    if (idx >= total) return;
    long long n = idx / 768;
    int c = (int)(idx - n * 768);
    float val;
    if (c < 192) {
        val = o[n * 192 + c];
    } else if (c < 480) {
        int c2 = c - 192;
        int pp = c2 / 3, comp = c2 - pp * 3;
        const float* fr = &f[n * 16];
        float gx = og[n * 288 + pp * 3 + 0] - fr[3];
        float gy = og[n * 288 + pp * 3 + 1] - fr[7];
        float gz = og[n * 288 + pp * 3 + 2] - fr[11];
        val = fr[0 * 4 + comp] * gx + fr[1 * 4 + comp] * gy + fr[2 * 4 + comp] * gz;
    } else if (c < 576) {
        int pp = c - 480;
        const float* fr = &f[n * 16];
        float gx = og[n * 288 + pp * 3 + 0] - fr[3];
        float gy = og[n * 288 + pp * 3 + 1] - fr[7];
        float gz = og[n * 288 + pp * 3 + 2] - fr[11];
        float l0 = fr[0] * gx + fr[4] * gy + fr[8]  * gz;
        float l1 = fr[1] * gx + fr[5] * gy + fr[9]  * gz;
        float l2 = fr[2] * gx + fr[6] * gy + fr[10] * gz;
        val = sqrtf(l0 * l0 + l1 * l1 + l2 * l2 + 1e-8f);
    } else {
        val = opair[n * 192 + (c - 576)];
    }
    os[idx] = (_Float16)val;
}

// ---------------------------------------------------------------------------
static inline int blocks_for(long long n, int t) { return (int)((n + t - 1) / t); }

extern "C" void kernel_launch(void* const* d_in, const int* in_sizes, int n_in,
                              void* d_out, int out_size, void* d_ws, size_t ws_size,
                              hipStream_t stream)
{
    const float* s    = (const float*)d_in[0];
    const float* z    = (const float*)d_in[1];
    const float* f    = (const float*)d_in[2];
    const int*   ei   = (const int*)  d_in[3];
    const float* W_[9]  = { (const float*)d_in[4],  (const float*)d_in[6],
                            (const float*)d_in[8],  (const float*)d_in[12],
                            (const float*)d_in[14], (const float*)d_in[16],
                            (const float*)d_in[10], (const float*)d_in[19],
                            (const float*)d_in[21] };
    const float* bq   = (const float*)d_in[5];
    const float* bk   = (const float*)d_in[7];
    const float* bv   = (const float*)d_in[9];
    const float* bb   = (const float*)d_in[11];
    const float* bqp  = (const float*)d_in[13];
    const float* bkp  = (const float*)d_in[15];
    const float* bvp  = (const float*)d_in[17];
    const float* hwts = (const float*)d_in[18];
    const float* bpair= (const float*)d_in[20];
    const float* bo   = (const float*)d_in[22];

    const int N = in_sizes[0] / 384;
    const int E = in_sizes[1] / 128;

    // ---- workspace layout ----
    float* ws = (float*)d_ws;
    size_t off = 0;
    float* q      = ws + off; off += (size_t)N * 192;
    float* kbuf   = ws + off; off += (size_t)N * 192;
    float* vbuf   = ws + off; off += (size_t)N * 192;
    float* qp     = ws + off; off += (size_t)N * 144;
    float* kp     = ws + off; off += (size_t)N * 144;
    float* vp     = ws + off; off += (size_t)N * 288;
    float* biasz  = ws + off; off += (size_t)E * 12;
    float* mbuf   = ws + off; off += (size_t)N * 12;
    float* denom  = ws + off; off += (size_t)N * 12;   // zero region starts here
    float* obuf   = ws + off; off += (size_t)N * 192;
    float* og     = ws + off; off += (size_t)N * 288;
    float* opair  = ws + off; off += (size_t)N * 192;
    float* attn   = ws + off; off += (size_t)E * 12;   // logits, then exp() in place
    float* vpair  = ws + off; off += (size_t)E * 192;
    // f16 region (all section sizes are multiples of 8 halves -> 16B aligned)
    _Float16* h16 = (_Float16*)(ws + off);
    size_t ho = 0;
    _Float16* sh   = h16 + ho; ho += (size_t)N * 384;
    _Float16* zh   = h16 + ho; ho += (size_t)E * 128;
    _Float16* osh  = h16 + ho; ho += (size_t)N * 768;
    const long long wsz[9] = { 192*384, 192*384, 192*384, 144*384, 144*384,
                               288*384, 12*128, 192*128, 384*768 };
    _Float16* wh[9];
    for (int w = 0; w < 9; ++w) { wh[w] = h16 + ho; ho += (size_t)wsz[w]; }
    (void)ws_size;

    dim3 blk(256);

    // ---- pre-convert GEMM operands to f16 ----
    ipa_cvt_f16<<<blocks_for((long long)N * 384 / 8, 256), blk, 0, stream>>>(s, sh, (long long)N * 384 / 8);
    ipa_cvt_f16<<<blocks_for((long long)E * 128 / 8, 256), blk, 0, stream>>>(z, zh, (long long)E * 128 / 8);
    for (int w = 0; w < 9; ++w)
        ipa_cvt_f16<<<blocks_for(wsz[w] / 8, 256), blk, 0, stream>>>(W_[w], wh[w], wsz[w] / 8);

    // ---- node projections (WMMA GEMMs) ----
    {
        dim3 g((N + GEMM_BM - 1) / GEMM_BM, (192 + GEMM_BN - 1) / GEMM_BN);
        ipa_gemm_wmma<<<g, blk, 0, stream>>>(sh, wh[0], bq, q,    N, 192, 384);
        ipa_gemm_wmma<<<g, blk, 0, stream>>>(sh, wh[1], bk, kbuf, N, 192, 384);
        ipa_gemm_wmma<<<g, blk, 0, stream>>>(sh, wh[2], bv, vbuf, N, 192, 384);
    }
    {
        dim3 g((N + GEMM_BM - 1) / GEMM_BM, (144 + GEMM_BN - 1) / GEMM_BN);
        ipa_gemm_wmma<<<g, blk, 0, stream>>>(sh, wh[3], bqp, qp, N, 144, 384);
        ipa_gemm_wmma<<<g, blk, 0, stream>>>(sh, wh[4], bkp, kp, N, 144, 384);
    }
    {
        dim3 g((N + GEMM_BM - 1) / GEMM_BM, (288 + GEMM_BN - 1) / GEMM_BN);
        ipa_gemm_wmma<<<g, blk, 0, stream>>>(sh, wh[5], bvp, vp, N, 288, 384);
    }

    // ---- rigid transforms to global frame ----
    ipa_pts_to_global<<<blocks_for((long long)N * 48, 256), blk, 0, stream>>>(f, qp, 48, (long long)N * 48);
    ipa_pts_to_global<<<blocks_for((long long)N * 48, 256), blk, 0, stream>>>(f, kp, 48, (long long)N * 48);
    ipa_pts_to_global<<<blocks_for((long long)N * 96, 256), blk, 0, stream>>>(f, vp, 96, (long long)N * 96);

    // ---- edge projections from z (WMMA GEMMs) ----
    {
        dim3 g((E + GEMM_BM - 1) / GEMM_BM, 1);
        ipa_gemm_wmma<<<g, blk, 0, stream>>>(zh, wh[6], bb, biasz, E, 12, 128);
    }
    {
        dim3 g((E + GEMM_BM - 1) / GEMM_BM, (192 + GEMM_BN - 1) / GEMM_BN);
        ipa_gemm_wmma<<<g, blk, 0, stream>>>(zh, wh[7], bpair, vpair, E, 192, 128);
    }

    // ---- init softmax / accumulator buffers ----
    ipa_fill<<<blocks_for((long long)N * 12, 256), blk, 0, stream>>>(mbuf, -INFINITY, (long long)N * 12);
    {
        long long nz = (long long)N * (12 + 192 + 288 + 192); // denom..opair contiguous
        ipa_fill<<<blocks_for(nz, 256), blk, 0, stream>>>(denom, 0.0f, nz);
    }

    // ---- edge logits + segment softmax ----
    int eh = E * 12;
    ipa_edge_logits<<<blocks_for(eh, 256), blk, 0, stream>>>(ei, q, kbuf, qp, kp, biasz, hwts, attn, E);
    ipa_seg_max   <<<blocks_for(eh, 256), blk, 0, stream>>>(ei, attn, mbuf, E);
    ipa_exp_denom <<<blocks_for(eh, 256), blk, 0, stream>>>(ei, attn, mbuf, denom, E);

    // ---- weighted scatter aggregation (8 contiguous channels per thread) ----
    {
        long long total = (long long)E * 84;
        ipa_aggregate<<<blocks_for(total, 256), blk, 0, stream>>>(
            ei, attn, denom, vbuf, vp, vpair, obuf, og, opair, total);
    }

    // ---- back to local frame, norms, concat (writes f16 o_s) ----
    ipa_finalize<<<blocks_for((long long)N * 768, 256), blk, 0, stream>>>(
        f, obuf, og, opair, osh, (long long)N * 768);

    // ---- output projection straight into d_out ----
    {
        dim3 g((N + GEMM_BM - 1) / GEMM_BM, (384 + GEMM_BN - 1) / GEMM_BN);
        ipa_gemm_wmma<<<g, blk, 0, stream>>>(osh, wh[8], bo, (float*)d_out, N, 384, 768);
    }

    // ---- second tuple output: z passthrough ----
    hipMemcpyAsync((float*)d_out + (size_t)N * 384, z,
                   (size_t)E * 128 * sizeof(float),
                   hipMemcpyDeviceToDevice, stream);
}
